// Seq2Seq_18897856102409
// MI455X (gfx1250) — matrix operand order repaired and checked by reading, not compiled, not run on hardware
//
#include <hip/hip_runtime.h>

// ---------------- problem dims ----------------
static constexpr int kV = 32000, kE = 100, kH = 128, kB = 64, kS = 64, kT = 64;
static constexpr int kG  = 4 * kH;   // 512 gate rows
static constexpr int kKX = 128;      // E padded to 128 (encoder input K)
static constexpr int kKD = 256;      // E+H=228 padded to 256 (decoder input K)
static constexpr int k2H = 256;      // concat(h, ctx)

typedef __attribute__((ext_vector_type(16))) __bf16 v16bf;
typedef __attribute__((ext_vector_type(8)))  __bf16 v8bf;
typedef __attribute__((ext_vector_type(8)))  float  v8f;

// Scheduling fence: stops LICM from hoisting loop-invariant weight-fragment
// loads out of the sequential timestep loops (which forced scratch spills).
#define SCHED_FENCE() asm volatile("" ::: "memory")

__device__ __forceinline__ v8f vzero() {
  v8f z = {0.f, 0.f, 0.f, 0.f, 0.f, 0.f, 0.f, 0.f};
  return z;
}

// A-matrix fragment (16x32 bf16, MxK), source row-major [M][ld]:
// lanes 0-15: row M=lane, K = kBase+{0..7, 16..23}; lanes 16-31: same rows, K+8.
__device__ __forceinline__ v16bf load_a_frag(const __bf16* base, int ld,
                                             int mBase, int kBase, int lane) {
  const int m  = mBase + (lane & 15);
  const int kh = (lane >> 4) << 3;  // 0 or 8
  const __bf16* p = base + m * ld + kBase + kh;
  v8bf lo = *(const v8bf*)(p);
  v8bf hi = *(const v8bf*)(p + 16);
  v16bf r;
#pragma unroll
  for (int j = 0; j < 8; ++j) { r[j] = lo[j]; r[j + 8] = hi[j]; }
  return r;
}

// B-matrix fragment (32x16 bf16, KxN), weights stored [N][ld] row-major (B = W^T):
// lanes 0-15: col N=lane, K = kBase+{0..7, 16..23}; lanes 16-31: same cols, K+8.
__device__ __forceinline__ v16bf load_b_frag(const __bf16* baseNK, int ld,
                                             int nBase, int kBase, int lane) {
  const int n  = nBase + (lane & 15);
  const int kh = (lane >> 4) << 3;  // 0 or 8
  const __bf16* p = baseNK + n * ld + kBase + kh;
  v8bf lo = *(const v8bf*)(p);
  v8bf hi = *(const v8bf*)(p + 16);
  v16bf r;
#pragma unroll
  for (int j = 0; j < 8; ++j) { r[j] = lo[j]; r[j + 8] = hi[j]; }
  return r;
}

__device__ __forceinline__ v8f wmma_bf(v16bf a, v16bf b, v8f c) {
  // (neg_a, A, neg_b, B, c_mod, C, reuse_a, reuse_b)
  return __builtin_amdgcn_wmma_f32_16x16x32_bf16(false, a, false, b, (short)0, c,
                                                 false, false);
}

__device__ __forceinline__ float sigm(float x) { return 1.0f / (1.0f + __expf(-x)); }

// ---------------- conversion / gather kernels ----------------
// dst[r][0..kDst) = bf16(src[r][kOff .. kOff+kCopy)) zero-padded to kDst
__global__ void cvt_slice_bf16(const float* __restrict__ src, __bf16* __restrict__ dst,
                               int rows, int ldSrc, int kOff, int kCopy, int kDst) {
  int i = blockIdx.x * blockDim.x + threadIdx.x;
  if (i >= rows * kDst) return;
  int r = i / kDst, k = i - r * kDst;
  float v = (k < kCopy) ? src[(size_t)r * ldSrc + kOff + k] : 0.0f;
  dst[i] = (__bf16)v;
}

// exB: [S][B][128] bf16, cols 0..99 = emb[src[b][s]], rest zero
__global__ void embed_src_kernel(const int* __restrict__ src, const float* __restrict__ emb,
                                 __bf16* __restrict__ exB) {
  int i = blockIdx.x * blockDim.x + threadIdx.x;
  if (i >= kS * kB * kKX) return;
  int k = i & (kKX - 1);
  int sb = i >> 7;
  int b = sb & (kB - 1);
  int s = sb >> 6;
  float v = 0.0f;
  if (k < kE) v = emb[(size_t)src[b * kS + s] * kE + k];
  exB[i] = (__bf16)v;
}

// etB: [T][B][256] bf16, cols 0..99 = emb[trg[b][t]], rest zero (ctx filled per step)
__global__ void embed_trg_kernel(const int* __restrict__ trg, const float* __restrict__ emb,
                                 __bf16* __restrict__ etB) {
  int i = blockIdx.x * blockDim.x + threadIdx.x;
  if (i >= kT * kB * kKD) return;
  int k = i & (kKD - 1);
  int tb = i >> 8;
  int b = tb & (kB - 1);
  int t = tb >> 6;
  float v = 0.0f;
  if (k < kE) v = emb[(size_t)trg[b * kT + t] * kE + k];
  etB[i] = (__bf16)v;
}

// ---------------- encoder: persistent single-workgroup LSTM ----------------
// 1024 threads = 32 waves. Wave (mTile = w&3, hTile = w>>2) owns the 16x16 block
// (batch rows mTile*16.., hidden cols hTile*16..) and computes all 4 gate tiles,
// so the LSTM nonlinearity runs entirely in registers. Weight fragments are
// re-fetched from L2 every K-chunk (SCHED_FENCE) instead of being hoisted+spilled.
__global__ void __launch_bounds__(1024)
encoder_kernel(const __bf16* __restrict__ exB,
               const __bf16* __restrict__ Wih, const __bf16* __restrict__ Whh,
               const float* __restrict__ bih, const float* __restrict__ bhh,
               float* __restrict__ enc_out,        // [B][S][H] f32
               __bf16* __restrict__ enc_out_bf,    // [B*S][H] bf16
               __bf16* __restrict__ hfin, float* __restrict__ cfin) {
  __shared__ __align__(16) float  c_lds[kB][kH];
  __shared__ __align__(16) __bf16 h_lds[kB][kH];

  const int tid = threadIdx.x;
  const int wave = tid >> 5, lane = tid & 31;
  const int mBase = (wave & 3) * 16;
  const int hBase = (wave >> 2) * 16;
  const int n16 = lane & 15, hi = lane >> 4;

  for (int i = tid; i < kB * kH; i += 1024) {
    (&c_lds[0][0])[i] = 0.0f;
    (&h_lds[0][0])[i] = (__bf16)0.0f;
  }
  __syncthreads();

  float bias[4];
#pragma unroll
  for (int g = 0; g < 4; ++g) {
    int n = g * kH + hBase + n16;
    bias[g] = bih[n] + bhh[n];
  }

  for (int s = 0; s < kS; ++s) {
    v8f acc[4] = {vzero(), vzero(), vzero(), vzero()};
    const __bf16* xa = exB + (size_t)s * kB * kKX;
#pragma unroll
    for (int kc = 0; kc < kKX / 32; ++kc) {  // x @ Wih^T  (K=128, E zero-padded)
      SCHED_FENCE();
      v16bf a = load_a_frag(xa, kKX, mBase, kc * 32, lane);
#pragma unroll
      for (int g = 0; g < 4; ++g)
        acc[g] = wmma_bf(a, load_b_frag(Wih, kKX, g * kH + hBase, kc * 32, lane), acc[g]);
    }
#pragma unroll
    for (int kc = 0; kc < kH / 32; ++kc) {   // h @ Whh^T  (K=128)
      SCHED_FENCE();
      v16bf a = load_a_frag(&h_lds[0][0], kH, mBase, kc * 32, lane);
#pragma unroll
      for (int g = 0; g < 4; ++g)
        acc[g] = wmma_bf(a, load_b_frag(Whh, kKX, g * kH + hBase, kc * 32, lane), acc[g]);
    }
    __syncthreads();  // all h reads done before h is rewritten
#pragma unroll
    for (int r = 0; r < 8; ++r) {
      int m = mBase + r + (hi << 3);
      int n = hBase + n16;
      float iv = sigm(acc[0][r] + bias[0]);
      float fv = sigm(acc[1][r] + bias[1]);
      float gv = tanhf(acc[2][r] + bias[2]);
      float ov = sigm(acc[3][r] + bias[3]);
      float cn = fv * c_lds[m][n] + iv * gv;
      float hn = ov * tanhf(cn);
      c_lds[m][n] = cn;
      h_lds[m][n] = (__bf16)hn;
      enc_out[((size_t)m * kS + s) * kH + n]    = hn;
      enc_out_bf[((size_t)m * kS + s) * kH + n] = (__bf16)hn;
    }
    __syncthreads();
  }
  for (int i = tid; i < kB * kH; i += 1024) {
    hfin[i] = (&h_lds[0][0])[i];
    cfin[i] = (&c_lds[0][0])[i];
  }
}

// ---------------- generic 64x64-tile GEMM: C = A @ B^T + bias ----------------
// A: [M][lda] bf16 row-major, Bw: [N][ldb] bf16 (weights), C: [M][ldc] f32.
__global__ void __launch_bounds__(256)
gemm_bias_kernel(const __bf16* __restrict__ A, int lda,
                 const __bf16* __restrict__ Bw, int ldb,
                 const float* __restrict__ bias,
                 float* __restrict__ C, int ldc, int kChunks) {
  const int tid = threadIdx.x, wave = tid >> 5, lane = tid & 31;
  const int mBase = blockIdx.y * 64 + (wave & 3) * 16;
  const int nBase0 = blockIdx.x * 64 + (wave >> 2) * 32;
  const int nBase1 = nBase0 + 16;
  const int n16 = lane & 15, hi = lane >> 4;
  v8f acc0 = vzero(), acc1 = vzero();
  for (int kc = 0; kc < kChunks; ++kc) {
    v16bf a = load_a_frag(A, lda, mBase, kc * 32, lane);
    acc0 = wmma_bf(a, load_b_frag(Bw, ldb, nBase0, kc * 32, lane), acc0);
    acc1 = wmma_bf(a, load_b_frag(Bw, ldb, nBase1, kc * 32, lane), acc1);
  }
  float b0 = bias[nBase0 + n16], b1 = bias[nBase1 + n16];
#pragma unroll
  for (int r = 0; r < 8; ++r) {
    int m = mBase + r + (hi << 3);
    C[(size_t)m * ldc + nBase0 + n16] = acc0[r] + b0;
    C[(size_t)m * ldc + nBase1 + n16] = acc1[r] + b1;
  }
}

// ---------------- decoder: persistent single-workgroup, attention + LSTM ----------------
__global__ void __launch_bounds__(1024)
decoder_kernel(const __bf16* __restrict__ etB,
               const __bf16* __restrict__ Wih, const __bf16* __restrict__ Whh,
               const float* __restrict__ bih, const float* __restrict__ bhh,
               const __bf16* __restrict__ WaB, const float* __restrict__ attnV,
               const float* __restrict__ encWb,   // [B*S][H], already + attn_b
               const float* __restrict__ enc_out, // [B][S][H] f32
               const __bf16* __restrict__ hinit, const float* __restrict__ cinit,
               __bf16* __restrict__ hcbf) {       // [T*B][256] bf16: [h | ctx]
  __shared__ __align__(16) float  c_lds[kB][kH];
  __shared__ __align__(16) __bf16 h_lds[kB][kH];
  __shared__ __align__(16) float  hWa[kB][kH];
  __shared__ __align__(16) float  sc[kB][kS];
  __shared__ __align__(16) __bf16 xbuf[kB][kKD];

  const int tid = threadIdx.x;
  const int wave = tid >> 5, lane = tid & 31;
  const int mBase = (wave & 3) * 16;
  const int hBase = (wave >> 2) * 16;
  const int n16 = lane & 15, hi = lane >> 4;

  for (int i = tid; i < kB * kH; i += 1024) {
    (&h_lds[0][0])[i] = hinit[i];
    (&c_lds[0][0])[i] = cinit[i];
  }
  __syncthreads();

  float bias[4];
#pragma unroll
  for (int g = 0; g < 4; ++g) {
    int n = g * kH + hBase + n16;
    bias[g] = bih[n] + bhh[n];
  }

  for (int t = 0; t < kT; ++t) {
    // P1: hWa = h @ Wa^T  (32 tiles: 4 m x 8 n, one per wave), + stage emb into xbuf
    {
      v8f acc = vzero();
#pragma unroll
      for (int kc = 0; kc < kH / 32; ++kc) {
        SCHED_FENCE();
        v16bf a = load_a_frag(&h_lds[0][0], kH, mBase, kc * 32, lane);
        acc = wmma_bf(a, load_b_frag(WaB, kH, (wave >> 2) * 16, kc * 32, lane), acc);
      }
#pragma unroll
      for (int r = 0; r < 8; ++r)
        hWa[mBase + r + (hi << 3)][(wave >> 2) * 16 + n16] = acc[r];
      for (int i = tid; i < kB * kKD; i += 1024)
        (&xbuf[0][0])[i] = etB[(size_t)t * kB * kKD + i];
    }
    __syncthreads();
    // P2: scores[b][s] = v . tanh(hWa[b] + encWb[b,s])  (wave -> b, lanes -> s)
#pragma unroll
    for (int bb = 0; bb < 2; ++bb) {
      int b = wave + bb * 32;
#pragma unroll
      for (int so = 0; so < 2; ++so) {
        int s = lane + so * 32;
        const float* ew = encWb + ((size_t)b * kS + s) * kH;
        float a = 0.0f;
        for (int h2 = 0; h2 < kH; ++h2)
          a += attnV[h2] * tanhf(hWa[b][h2] + ew[h2]);
        sc[b][s] = a;
      }
    }
    __syncthreads();
    // P3: softmax over s per batch row
    if (tid < kB) {
      int b = tid;
      float mx = -1e30f;
      for (int s2 = 0; s2 < kS; ++s2) mx = fmaxf(mx, sc[b][s2]);
      float sum = 0.0f;
      for (int s2 = 0; s2 < kS; ++s2) { float e = __expf(sc[b][s2] - mx); sc[b][s2] = e; sum += e; }
      float inv = 1.0f / sum;
      for (int s2 = 0; s2 < kS; ++s2) sc[b][s2] *= inv;
    }
    __syncthreads();
    // P4: ctx[b][h] = sum_s w[s] * enc_out[b][s][h]; write into xbuf cols 100..227 + hc staging
    for (int p = tid; p < kB * kH; p += 1024) {
      int b = p >> 7, h2 = p & (kH - 1);
      const float* eo = enc_out + (size_t)b * kS * kH + h2;
      float a = 0.0f;
      for (int s2 = 0; s2 < kS; ++s2) a += sc[b][s2] * eo[(size_t)s2 * kH];
      __bf16 cb = (__bf16)a;
      xbuf[b][kE + h2] = cb;
      hcbf[((size_t)t * kB + b) * k2H + kH + h2] = cb;
    }
    __syncthreads();
    // P5: LSTM cell: gates = x @ Wih^T (K=256) + h @ Whh^T (K=128)
    v8f acc[4] = {vzero(), vzero(), vzero(), vzero()};
#pragma unroll
    for (int kc = 0; kc < kKD / 32; ++kc) {
      SCHED_FENCE();
      v16bf a = load_a_frag(&xbuf[0][0], kKD, mBase, kc * 32, lane);
#pragma unroll
      for (int g = 0; g < 4; ++g)
        acc[g] = wmma_bf(a, load_b_frag(Wih, kKD, g * kH + hBase, kc * 32, lane), acc[g]);
    }
#pragma unroll
    for (int kc = 0; kc < kH / 32; ++kc) {
      SCHED_FENCE();
      v16bf a = load_a_frag(&h_lds[0][0], kH, mBase, kc * 32, lane);
#pragma unroll
      for (int g = 0; g < 4; ++g)
        acc[g] = wmma_bf(a, load_b_frag(Whh, kKX, g * kH + hBase, kc * 32, lane), acc[g]);
    }
    __syncthreads();
#pragma unroll
    for (int r = 0; r < 8; ++r) {
      int m = mBase + r + (hi << 3);
      int n = hBase + n16;
      float iv = sigm(acc[0][r] + bias[0]);
      float fv = sigm(acc[1][r] + bias[1]);
      float gv = tanhf(acc[2][r] + bias[2]);
      float ov = sigm(acc[3][r] + bias[3]);
      float cn = fv * c_lds[m][n] + iv * gv;
      float hn = ov * tanhf(cn);
      c_lds[m][n] = cn;
      h_lds[m][n] = (__bf16)hn;
      hcbf[((size_t)t * kB + m) * k2H + n] = (__bf16)hn;
    }
    __syncthreads();
  }
}

// ---------------- batched FC head: out[b][t][v] = hc[t*B+b] . fc_W[v] + fc_b[v] ----------------
// M = T*B = 4096, N = V = 32000, K = 256. fc_W bf16 (16.4 MB) is L2-resident.
__global__ void __launch_bounds__(256)
fc_kernel(const __bf16* __restrict__ hc, const __bf16* __restrict__ Wfc,
          const float* __restrict__ bfc, float* __restrict__ out) {
  const int tid = threadIdx.x, wave = tid >> 5, lane = tid & 31;
  const int mBase = blockIdx.y * 64 + (wave & 3) * 16;
  const int nBase0 = blockIdx.x * 64 + (wave >> 2) * 32;
  const int nBase1 = nBase0 + 16;
  const int n16 = lane & 15, hi = lane >> 4;
  v8f acc0 = vzero(), acc1 = vzero();
#pragma unroll
  for (int kc = 0; kc < k2H / 32; ++kc) {
    // keep the pipeline fed: prefetch next chunk of the A row this lane reads
    if (kc + 1 < k2H / 32)
      __builtin_prefetch(hc + (size_t)(mBase + (lane & 15)) * k2H + (kc + 1) * 32, 0, 3);
    v16bf a = load_a_frag(hc, k2H, mBase, kc * 32, lane);
    acc0 = wmma_bf(a, load_b_frag(Wfc, k2H, nBase0, kc * 32, lane), acc0);
    acc1 = wmma_bf(a, load_b_frag(Wfc, k2H, nBase1, kc * 32, lane), acc1);
  }
  float b0 = bfc[nBase0 + n16], b1 = bfc[nBase1 + n16];
#pragma unroll
  for (int r = 0; r < 8; ++r) {
    int m = mBase + r + (hi << 3);
    int t = m >> 6, b = m & (kB - 1);
    float* orow = out + ((size_t)b * kT + t) * kV;
    orow[nBase0 + n16] = acc0[r] + b0;
    orow[nBase1 + n16] = acc1[r] + b1;
  }
}

// ---------------- host launcher ----------------
extern "C" void kernel_launch(void* const* d_in, const int* in_sizes, int n_in,
                              void* d_out, int out_size, void* d_ws, size_t ws_size,
                              hipStream_t stream) {
  (void)in_sizes; (void)n_in; (void)out_size; (void)ws_size;
  const int*   src    = (const int*)  d_in[0];
  const int*   trg    = (const int*)  d_in[1];
  const float* emb    = (const float*)d_in[2];
  const float* encWih = (const float*)d_in[3];
  const float* encWhh = (const float*)d_in[4];
  const float* encBih = (const float*)d_in[5];
  const float* encBhh = (const float*)d_in[6];
  const float* decWih = (const float*)d_in[7];
  const float* decWhh = (const float*)d_in[8];
  const float* decBih = (const float*)d_in[9];
  const float* decBhh = (const float*)d_in[10];
  const float* attnW  = (const float*)d_in[11];
  const float* attnB  = (const float*)d_in[12];
  const float* attnV  = (const float*)d_in[13];
  const float* fcW    = (const float*)d_in[14];
  const float* fcB    = (const float*)d_in[15];
  float* out = (float*)d_out;

  char* ws = (char*)d_ws;
  size_t off = 0;
  auto take = [&](size_t bytes) -> char* {
    char* p = ws + off;
    off += (bytes + 255) & ~(size_t)255;
    return p;
  };
  __bf16* encWihB   = (__bf16*)take((size_t)kG * kKX * 2);
  __bf16* encWhhB   = (__bf16*)take((size_t)kG * kKX * 2);
  __bf16* decWihB   = (__bf16*)take((size_t)kG * kKD * 2);
  __bf16* decWhhB   = (__bf16*)take((size_t)kG * kKX * 2);
  __bf16* WaB       = (__bf16*)take((size_t)kH * kH * 2);
  __bf16* WbB       = (__bf16*)take((size_t)kH * kH * 2);
  __bf16* fcWB      = (__bf16*)take((size_t)kV * k2H * 2);
  __bf16* exB       = (__bf16*)take((size_t)kS * kB * kKX * 2);
  __bf16* etB       = (__bf16*)take((size_t)kT * kB * kKD * 2);
  float*  enc_out   = (float*) take((size_t)kB * kS * kH * 4);
  __bf16* enc_out_b = (__bf16*)take((size_t)kB * kS * kH * 2);
  float*  encWb     = (float*) take((size_t)kB * kS * kH * 4);
  __bf16* hcbf      = (__bf16*)take((size_t)kT * kB * k2H * 2);
  __bf16* hfin      = (__bf16*)take((size_t)kB * kH * 2);
  float*  cfin      = (float*) take((size_t)kB * kH * 4);

  const int THR = 256;
  auto cdiv = [](int a, int b) { return (a + b - 1) / b; };

  // weight conversions (fp32 -> bf16, zero-padded K)
  cvt_slice_bf16<<<cdiv(kG * kKX, THR), THR, 0, stream>>>(encWih, encWihB, kG, kE, 0, kE, kKX);
  cvt_slice_bf16<<<cdiv(kG * kKX, THR), THR, 0, stream>>>(encWhh, encWhhB, kG, kH, 0, kH, kKX);
  cvt_slice_bf16<<<cdiv(kG * kKD, THR), THR, 0, stream>>>(decWih, decWihB, kG, kE + kH, 0, kE + kH, kKD);
  cvt_slice_bf16<<<cdiv(kG * kKX, THR), THR, 0, stream>>>(decWhh, decWhhB, kG, kH, 0, kH, kKX);
  cvt_slice_bf16<<<cdiv(kH * kH, THR), THR, 0, stream>>>(attnW, WaB, kH, k2H, 0, kH, kH);
  cvt_slice_bf16<<<cdiv(kH * kH, THR), THR, 0, stream>>>(attnW, WbB, kH, k2H, kH, kH, kH);
  cvt_slice_bf16<<<cdiv(kV * k2H, THR), THR, 0, stream>>>(fcW, fcWB, kV, k2H, 0, k2H, k2H);

  // embeddings
  embed_src_kernel<<<cdiv(kS * kB * kKX, THR), THR, 0, stream>>>(src, emb, exB);
  embed_trg_kernel<<<cdiv(kT * kB * kKD, THR), THR, 0, stream>>>(trg, emb, etB);

  // encoder (sequential, persistent workgroup)
  encoder_kernel<<<1, 1024, 0, stream>>>(exB, encWihB, encWhhB, encBih, encBhh,
                                         enc_out, enc_out_b, hfin, cfin);

  // attention precompute: encWb = enc_out @ Wb^T + attn_b   (M=4096, N=128, K=128)
  gemm_bias_kernel<<<dim3(kH / 64, (kB * kS) / 64), 256, 0, stream>>>(
      enc_out_b, kH, WbB, kH, attnB, encWb, kH, kH / 32);

  // decoder (sequential, persistent workgroup)
  decoder_kernel<<<1, 1024, 0, stream>>>(etB, decWihB, decWhhB, decBih, decBhh,
                                         WaB, attnV, encWb, enc_out, hfin, cfin, hcbf);

  // batched FC head over all T steps: (4096 x 32000 x 256)
  fc_kernel<<<dim3(kV / 64, (kT * kB) / 64), 256, 0, stream>>>(hcbf, fcWB, fcB, out);
}
